// GCN_90177133346981
// MI455X (gfx1250) — matrix operand order repaired and checked by reading, not compiled, not run on hardware
//
#include <hip/hip_runtime.h>

// ---------------------------------------------------------------------------
// 2-layer GCN for gfx1250 (MI455X): bf16 WMMA GEMMs with register-resident
// B-fragments (weights loaded from transposed-LDS once per wave), grid-stride
// M-tile loop, fused dinv scaling, L2-resident atomic scatter aggregation.
// ---------------------------------------------------------------------------

typedef __attribute__((ext_vector_type(16))) __bf16 v16bf;
typedef __attribute__((ext_vector_type(8)))  float  v8f;

#define N_NODES 100000
#define N_EDGES 1600000
#define IN_CH   128
#define HID_CH  64
#define OUT_CH  40
#define OUT_PAD 48
#define M_TILES ((N_NODES + 15) / 16)   // 6250 exactly
#define GEMM_BLOCKS 391                 // ~2 tiles per wave (391*8*2 >= 6250)

// ---------------- degree / dinv -------------------------------------------

__global__ void k_init_deg(float* __restrict__ deg) {
  int i = blockIdx.x * blockDim.x + threadIdx.x;
  if (i < N_NODES) deg[i] = 1.0f;            // self-loop contributes 1
}

__global__ void k_count_deg(const int* __restrict__ ei, float* __restrict__ deg) {
  int e = blockIdx.x * blockDim.x + threadIdx.x;
  if (e < N_EDGES) atomicAdd(&deg[ei[N_EDGES + e]], 1.0f);   // targets = row 1
}

__global__ void k_dinv(float* __restrict__ deg_dinv) {
  int i = blockIdx.x * blockDim.x + threadIdx.x;
  if (i < N_NODES) deg_dinv[i] = rsqrtf(deg_dinv[i]);        // deg >= 1 always
}

// ---------------- layer 1 GEMM: hs1 = dinv * (x @ W1); agg1 = hs1 ----------
// W1 transposed into LDS once per block; each wave hoists all 16 B-fragments
// into registers, then grid-strides over M tiles with pure A-stream + WMMA.

__global__ void __launch_bounds__(256)
k_gemm1(const float* __restrict__ x, const float* __restrict__ W1,
        const float* __restrict__ dinv,
        float* __restrict__ hs1, float* __restrict__ agg1) {
  __shared__ __align__(32) __bf16 sWt[HID_CH * IN_CH];   // 64 x 128, 16 KB
  for (int j = threadIdx.x; j < HID_CH * IN_CH; j += 256) {
    const int n = j >> 7;            // output channel
    const int k = j & 127;           // input channel
    sWt[j] = (__bf16)W1[k * HID_CH + n];
  }
  __syncthreads();

  const int  wave = threadIdx.x >> 5;        // wave32
  const int  lane = threadIdx.x & 31;
  const int  lrow  = lane & 15;
  const bool hihalf = lane >= 16;
  const int  nwaves = gridDim.x * 8;

  // B fragments (32x16 bf16 each): lane%16 = column, lane/16 selects K-half.
  // All (ks, n) fragments live in registers for the whole kernel.
  v16bf barr[4][4];
  #pragma unroll
  for (int ks = 0; ks < 4; ++ks) {
    const int kb = ks * 32 + (hihalf ? 16 : 0);
    #pragma unroll
    for (int n = 0; n < 4; ++n)
      barr[ks][n] = *reinterpret_cast<const v16bf*>(
          &sWt[(n * 16 + lrow) * IN_CH + kb]);
  }

  for (int tile = blockIdx.x * 8 + wave; tile < M_TILES; tile += nwaves) {
    const int mbase = tile * 16;
    v8f acc[4] = {};
    #pragma unroll
    for (int ks = 0; ks < 4; ++ks) {          // K = 128 in steps of 32
      // A fragment: lane<16 -> K offsets {0..15}, lane>=16 -> {16..31}
      const float* xp = x + (size_t)(mbase + lrow) * IN_CH + ks * 32 + (hihalf ? 16 : 0);
      v16bf a;
      #pragma unroll
      for (int j = 0; j < 8; ++j) {
        a[j]     = (__bf16)xp[j];
        a[j + 8] = (__bf16)xp[j + 8];
      }
      #pragma unroll
      for (int n = 0; n < 4; ++n)
        acc[n] = __builtin_amdgcn_wmma_f32_16x16x32_bf16(
            false, a, false, barr[ks][n], (short)0, acc[n], false, false);
    }

    // C/D layout: element v -> row mbase + v + (lane>=16 ? 8 : 0), col = n*16 + lane%16
    #pragma unroll
    for (int v = 0; v < 8; ++v) {
      const int   r = mbase + (hihalf ? 8 : 0) + v;
      const float d = dinv[r];
      #pragma unroll
      for (int n = 0; n < 4; ++n) {
        const float  val = acc[n][v] * d;
        const size_t idx = (size_t)r * HID_CH + n * 16 + lrow;
        hs1[idx]  = val;    // pre-scaled features for edge gather
        agg1[idx] = val;    // self-loop seed
      }
    }
  }
}

// ---------------- edge scatter, layer 1 (64 ch, 16 lanes/edge x float4) ----

__global__ void k_scatter1(const int* __restrict__ ei,
                           const float* __restrict__ hs1,
                           float* __restrict__ agg1) {
  long long t = (long long)blockIdx.x * blockDim.x + threadIdx.x;
  if (t >= (long long)N_EDGES * 16) return;
  const int e = (int)(t >> 4);
  const int g = (int)(t & 15);
  __builtin_prefetch(ei + e + 8192, 0, 0);            // global_prefetch_b8
  const int row = ei[e];
  const int col = ei[N_EDGES + e];
  const float4 v = *(const float4*)(hs1 + (size_t)row * HID_CH + g * 4);
  float* dst = agg1 + (size_t)col * HID_CH + g * 4;
  atomicAdd(dst + 0, v.x);
  atomicAdd(dst + 1, v.y);
  atomicAdd(dst + 2, v.z);
  atomicAdd(dst + 3, v.w);
}

// ---- layer 2 GEMM: z = relu(dinv*agg1 + b1); hs2 = dinv*(z @ W2); agg2 = hs2

__global__ void __launch_bounds__(256)
k_gemm2(const float* __restrict__ agg1, const float* __restrict__ dinv,
        const float* __restrict__ b1, const float* __restrict__ W2,
        float* __restrict__ hs2, float* __restrict__ agg2) {
  __shared__ __align__(32) __bf16 sWt[OUT_PAD * HID_CH]; // 48 x 64, cols 40..47 zero
  for (int j = threadIdx.x; j < OUT_PAD * HID_CH; j += 256) {
    const int n = j >> 6;            // output channel (padded)
    const int k = j & 63;            // hidden channel
    sWt[j] = (__bf16)((n < OUT_CH) ? W2[k * OUT_CH + n] : 0.0f);
  }
  __syncthreads();

  const int  wave = threadIdx.x >> 5;
  const int  lane = threadIdx.x & 31;
  const int  lrow  = lane & 15;
  const bool hihalf = lane >= 16;
  const int  nwaves = gridDim.x * 8;

  v16bf barr[2][3];
  #pragma unroll
  for (int ks = 0; ks < 2; ++ks) {
    const int kb = ks * 32 + (hihalf ? 16 : 0);
    #pragma unroll
    for (int n = 0; n < 3; ++n)
      barr[ks][n] = *reinterpret_cast<const v16bf*>(
          &sWt[(n * 16 + lrow) * HID_CH + kb]);
  }

  // b1 slices this lane needs (channels cb..cb+7 and cb+8..cb+15 per ks)
  for (int tile = blockIdx.x * 8 + wave; tile < M_TILES; tile += nwaves) {
    const int   mbase = tile * 16;
    const int   r_a   = mbase + lrow;
    const float d_a   = dinv[r_a];
    const float* ap   = agg1 + (size_t)r_a * HID_CH;

    v8f acc[3] = {};
    #pragma unroll
    for (int ks = 0; ks < 2; ++ks) {          // K = 64 in steps of 32
      const int cb = ks * 32 + (hihalf ? 16 : 0);
      v16bf a;                                // fused relu(dinv*agg1 + b1) -> bf16
      #pragma unroll
      for (int j = 0; j < 8; ++j) {
        float z0 = d_a * ap[cb + j]      + b1[cb + j];
        float z1 = d_a * ap[cb + 8 + j] + b1[cb + 8 + j];
        a[j]     = (__bf16)(z0 > 0.0f ? z0 : 0.0f);
        a[j + 8] = (__bf16)(z1 > 0.0f ? z1 : 0.0f);
      }
      #pragma unroll
      for (int n = 0; n < 3; ++n)
        acc[n] = __builtin_amdgcn_wmma_f32_16x16x32_bf16(
            false, a, false, barr[ks][n], (short)0, acc[n], false, false);
    }

    #pragma unroll
    for (int v = 0; v < 8; ++v) {
      const int   r = mbase + (hihalf ? 8 : 0) + v;
      const float d = dinv[r];
      #pragma unroll
      for (int n = 0; n < 3; ++n) {
        const int col = n * 16 + lrow;
        if (col < OUT_CH) {
          const float  val = acc[n][v] * d;
          const size_t idx = (size_t)r * OUT_CH + col;
          hs2[idx]  = val;
          agg2[idx] = val;
        }
      }
    }
  }
}

// ---------------- edge scatter, layer 2 (40 ch, 10 float4 groups/edge) -----

__global__ void k_scatter2(const int* __restrict__ ei,
                           const float* __restrict__ hs2,
                           float* __restrict__ agg2) {
  long long t = (long long)blockIdx.x * blockDim.x + threadIdx.x;
  if (t >= (long long)N_EDGES * 10) return;
  const int e = (int)(t / 10);
  const int g = (int)(t % 10);
  const int row = ei[e];
  const int col = ei[N_EDGES + e];
  const float4 v = *(const float4*)(hs2 + (size_t)row * OUT_CH + g * 4);
  float* dst = agg2 + (size_t)col * OUT_CH + g * 4;
  atomicAdd(dst + 0, v.x);
  atomicAdd(dst + 1, v.y);
  atomicAdd(dst + 2, v.z);
  atomicAdd(dst + 3, v.w);
}

// ---------------- finalize: out = dinv * agg2 + b2 -------------------------

__global__ void k_final(const float* __restrict__ agg2,
                        const float* __restrict__ dinv,
                        const float* __restrict__ b2,
                        float* __restrict__ out) {
  long long t = (long long)blockIdx.x * blockDim.x + threadIdx.x;
  if (t >= (long long)N_NODES * OUT_CH) return;
  const int i = (int)(t / OUT_CH);
  const int c = (int)(t % OUT_CH);
  out[t] = dinv[i] * agg2[t] + b2[c];
}

// ---------------------------------------------------------------------------

extern "C" void kernel_launch(void* const* d_in, const int* in_sizes, int n_in,
                              void* d_out, int out_size, void* d_ws, size_t ws_size,
                              hipStream_t stream) {
  const float* x  = (const float*)d_in[0];
  const int*   ei = (const int*)d_in[1];       // [2, E] (int32 per harness)
  const float* W1 = (const float*)d_in[2];
  const float* b1 = (const float*)d_in[3];
  const float* W2 = (const float*)d_in[4];
  const float* b2 = (const float*)d_in[5];
  float* out = (float*)d_out;

  char* ws = (char*)d_ws;
  size_t off = 0;
  auto carve = [&](size_t bytes) {
    char* p = ws + off;
    off = (off + bytes + 255) & ~(size_t)255;
    return p;
  };
  float* dinv = (float*)carve((size_t)N_NODES * sizeof(float));          // deg -> dinv in place
  float* hs1  = (float*)carve((size_t)N_NODES * HID_CH * sizeof(float));
  float* agg1 = (float*)carve((size_t)N_NODES * HID_CH * sizeof(float));
  float* hs2  = (float*)carve((size_t)N_NODES * OUT_CH * sizeof(float));
  float* agg2 = (float*)carve((size_t)N_NODES * OUT_CH * sizeof(float));

  const int T = 256;
  // degrees
  k_init_deg <<<(N_NODES + T - 1) / T, T, 0, stream>>>(dinv);
  k_count_deg<<<(N_EDGES + T - 1) / T, T, 0, stream>>>(ei, dinv);
  k_dinv     <<<(N_NODES + T - 1) / T, T, 0, stream>>>(dinv);
  // layer 1
  k_gemm1    <<<GEMM_BLOCKS, T, 0, stream>>>(x, W1, dinv, hs1, agg1);
  k_scatter1 <<<(int)(((long long)N_EDGES * 16 + T - 1) / T), T, 0, stream>>>(ei, hs1, agg1);
  // layer 2 (relu + bias + dinv fused into A-fragment build)
  k_gemm2    <<<GEMM_BLOCKS, T, 0, stream>>>(agg1, dinv, b1, W2, hs2, agg2);
  k_scatter2 <<<(int)(((long long)N_EDGES * 10 + T - 1) / T), T, 0, stream>>>(ei, hs2, agg2);
  // finalize
  k_final    <<<(int)(((long long)N_NODES * OUT_CH + T - 1) / T), T, 0, stream>>>(agg2, dinv, b2, out);
  (void)in_sizes; (void)n_in; (void)out_size; (void)ws_size;
}
